// Model_34488587387189
// MI455X (gfx1250) — matrix operand order repaired and checked by reading, not compiled, not run on hardware
//
#include <hip/hip_runtime.h>
#include <hip/hip_bf16.h>
#include <math.h>

#define BB 2048
#define DD 1024
#define KK 128
#define INV_TEMP 20.0f   // 1 / 0.05

typedef __attribute__((ext_vector_type(16))) _Float16 v16h;
typedef __attribute__((ext_vector_type(8)))  _Float16 v8h;
typedef __attribute__((ext_vector_type(8)))  float    v8f;

#define CAT16(lo, hi) __builtin_shufflevector((lo), (hi), 0,1,2,3,4,5,6,7,8,9,10,11,12,13,14,15)

static __device__ __forceinline__ v8f wmma_f16(v16h a, v16h b, v8f c) {
  // (neg_a, A, neg_b, B, c_mod, C, reuse_a, reuse_b)
  return __builtin_amdgcn_wmma_f32_16x16x32_f16(false, a, false, b, (short)0, c, false, false);
}

// ---------------------------------------------------------------------------
// Kernel 1: L2-normalize each row of feat and neg; emit split-f16 (hi + lo).
// One 256-thread block per row; each thread owns 4 contiguous floats (float4).
// ---------------------------------------------------------------------------
__global__ __launch_bounds__(256) void normalize_split(
    const float* __restrict__ feat, const float* __restrict__ neg,
    _Float16* __restrict__ fh, _Float16* __restrict__ fl,
    _Float16* __restrict__ nh, _Float16* __restrict__ nl)
{
  __shared__ float sred[2][8];
  const int r    = blockIdx.x;
  const int t    = threadIdx.x;
  const int lane = t & 31;
  const int wave = t >> 5;

  const float4 fv = ((const float4*)(feat + (size_t)r * DD))[t];
  const float4 nv = ((const float4*)(neg  + (size_t)r * DD))[t];
  float fs = fv.x*fv.x + fv.y*fv.y + fv.z*fv.z + fv.w*fv.w;
  float ns = nv.x*nv.x + nv.y*nv.y + nv.z*nv.z + nv.w*nv.w;

  #pragma unroll
  for (int m = 16; m > 0; m >>= 1) {
    fs += __shfl_xor(fs, m, 32);
    ns += __shfl_xor(ns, m, 32);
  }
  if (lane == 0) { sred[0][wave] = fs; sred[1][wave] = ns; }
  __syncthreads();
  if (t == 0) {
    float a = 0.f, b = 0.f;
    #pragma unroll
    for (int w = 0; w < 8; ++w) { a += sred[0][w]; b += sred[1][w]; }
    sred[0][0] = 1.0f / fmaxf(sqrtf(a), 1e-12f);
    sred[1][0] = 1.0f / fmaxf(sqrtf(b), 1e-12f);
  }
  __syncthreads();
  const float fscale = sred[0][0];
  const float nscale = sred[1][0];

  const size_t base = (size_t)r * DD + (size_t)t * 4;
  const float fx[4] = {fv.x, fv.y, fv.z, fv.w};
  const float nx[4] = {nv.x, nv.y, nv.z, nv.w};
  #pragma unroll
  for (int e = 0; e < 4; ++e) {
    float y = fx[e] * fscale;
    _Float16 h = (_Float16)y;
    fh[base + e] = h;
    fl[base + e] = (_Float16)(y - (float)h);
    y = nx[e] * nscale;
    h = (_Float16)y;
    nh[base + e] = h;
    nl[base + e] = (_Float16)(y - (float)h);
  }
}

// ---------------------------------------------------------------------------
// Kernel 2: S = f . n^T via v_wmma_f32_16x16x32_f16, split-f16 compensation:
//   S ~= Ah*Bh + Ah*Bl + Al*Bh   (fp32-class accuracy on f16 matrix pipes)
// One wave per 16x32 output strip (2 N-tiles reuse the A fragment).
// A fragment (ISA 7.12.2, 16-bit A 16x32): lane L -> row (L&15);
//   hi = L>>4 selects K = hi*16 .. hi*16+15 contiguous (same order as B; not the ISA table order).
// B fragment (32x16): lane L -> column (L&15); K = (L>>4)*16 .. +15 contiguous.
// D (16x16 f32): lane L, VGPR v -> row (L>>4)*8 + v, col (L&15).
// ---------------------------------------------------------------------------
__global__ __launch_bounds__(256) void sim_gemm(
    const _Float16* __restrict__ fh, const _Float16* __restrict__ fl,
    const _Float16* __restrict__ nh, const _Float16* __restrict__ nl,
    float* __restrict__ S)
{
  const int lane = threadIdx.x & 31;
  const int wave = threadIdx.x >> 5;
  const int m16  = lane & 15;
  const int hi   = lane >> 4;
  const int i0   = blockIdx.x * 16;
  const int j0   = (blockIdx.y * 8 + wave) * 32;

  const size_t arow  = (size_t)(i0 + m16) * DD;       // A: row of f
  const size_t brow0 = (size_t)(j0 + m16) * DD;       // B tile 0: row of n = column of S
  const size_t brow1 = (size_t)(j0 + 16 + m16) * DD;  // B tile 1

  v8f acc0 = {};
  v8f acc1 = {};

  for (int k0 = 0; k0 < DD; k0 += 32) {
    const int ka = k0 + hi * 16;   // A chunk ka .. ka+15
    const int kb = k0 + hi * 16;   // B chunk kb .. kb+15

    v16h Ah = CAT16(*(const v8h*)(fh + arow + ka), *(const v8h*)(fh + arow + ka + 8));
    v16h Al = CAT16(*(const v8h*)(fl + arow + ka), *(const v8h*)(fl + arow + ka + 8));

    v16h B0h = CAT16(*(const v8h*)(nh + brow0 + kb), *(const v8h*)(nh + brow0 + kb + 8));
    v16h B0l = CAT16(*(const v8h*)(nl + brow0 + kb), *(const v8h*)(nl + brow0 + kb + 8));
    v16h B1h = CAT16(*(const v8h*)(nh + brow1 + kb), *(const v8h*)(nh + brow1 + kb + 8));
    v16h B1l = CAT16(*(const v8h*)(nl + brow1 + kb), *(const v8h*)(nl + brow1 + kb + 8));

    // Interleave the two accumulator chains for WMMA latency hiding.
    acc0 = wmma_f16(Ah, B0h, acc0);
    acc1 = wmma_f16(Ah, B1h, acc1);
    acc0 = wmma_f16(Ah, B0l, acc0);
    acc1 = wmma_f16(Ah, B1l, acc1);
    acc0 = wmma_f16(Al, B0h, acc0);
    acc1 = wmma_f16(Al, B1h, acc1);
  }

  float* srow = S + (size_t)(i0 + hi * 8) * BB + j0 + m16;
  #pragma unroll
  for (int v = 0; v < 8; ++v) {
    srow[(size_t)v * BB]      = acc0[v];
    srow[(size_t)v * BB + 16] = acc1[v];
  }
}

// ---------------------------------------------------------------------------
// Kernel 3: per-row logsumexp over {pos, 128 gathered negatives}.
// One 128-thread block (4 waves) per row; thread t owns negative t.
// ---------------------------------------------------------------------------
__global__ __launch_bounds__(128) void row_loss(
    const float* __restrict__ S, const int* __restrict__ neg_idx,
    float* __restrict__ partial)
{
  __shared__ float red[4];
  const int i    = blockIdx.x;
  const int t    = threadIdx.x;
  const int lane = t & 31;
  const int wave = t >> 5;

  const float pos = S[(size_t)i * BB + i] * INV_TEMP;
  const int   j   = neg_idx[i * KK + t];
  const float l   = S[(size_t)i * BB + j] * INV_TEMP;

  // block max
  float m = l;
  #pragma unroll
  for (int d = 16; d > 0; d >>= 1) m = fmaxf(m, __shfl_xor(m, d, 32));
  if (lane == 0) red[wave] = m;
  __syncthreads();
  m = fmaxf(fmaxf(red[0], red[1]), fmaxf(red[2], red[3]));
  m = fmaxf(m, pos);
  __syncthreads();

  // block sum of exp
  float e = __expf(l - m);
  #pragma unroll
  for (int d = 16; d > 0; d >>= 1) e += __shfl_xor(e, d, 32);
  if (lane == 0) red[wave] = e;
  __syncthreads();
  if (t == 0) {
    const float s = red[0] + red[1] + red[2] + red[3] + __expf(pos - m);
    partial[i] = m + __logf(s) - pos;   // logsumexp - logits[:,0]
  }
}

// ---------------------------------------------------------------------------
// Kernel 4: mean over 2048 per-row partials -> scalar loss.
// ---------------------------------------------------------------------------
__global__ __launch_bounds__(256) void final_reduce(
    const float* __restrict__ partial, float* __restrict__ out)
{
  __shared__ float red[8];
  const int t    = threadIdx.x;
  const int lane = t & 31;
  const int wave = t >> 5;
  float s = 0.f;
  for (int i = t; i < BB; i += 256) s += partial[i];
  #pragma unroll
  for (int d = 16; d > 0; d >>= 1) s += __shfl_xor(s, d, 32);
  if (lane == 0) red[wave] = s;
  __syncthreads();
  if (t == 0) {
    float tot = 0.f;
    #pragma unroll
    for (int w = 0; w < 8; ++w) tot += red[w];
    out[0] = tot / (float)BB;
  }
}

extern "C" void kernel_launch(void* const* d_in, const int* in_sizes, int n_in,
                              void* d_out, int out_size, void* d_ws, size_t ws_size,
                              hipStream_t stream) {
  (void)in_sizes; (void)n_in; (void)out_size; (void)ws_size;
  const float* feat    = (const float*)d_in[0];
  const float* neg     = (const float*)d_in[1];
  // d_in[2] = aim_labels: only used host-side in the reference's sampling.
  const int*   neg_idx = (const int*)d_in[3];
  float*       out     = (float*)d_out;

  // Workspace layout (all 16B-aligned, ~32 MiB total; L2-resident on MI455X):
  char* w = (char*)d_ws;
  const size_t HALF_MAT = (size_t)BB * DD * sizeof(_Float16);  // 4 MiB
  _Float16* fh = (_Float16*)(w);
  _Float16* fl = (_Float16*)(w + HALF_MAT);
  _Float16* nh = (_Float16*)(w + 2 * HALF_MAT);
  _Float16* nl = (_Float16*)(w + 3 * HALF_MAT);
  float*    S  = (float*)   (w + 4 * HALF_MAT);                // 16 MiB
  float*    pp = (float*)   (w + 4 * HALF_MAT + (size_t)BB * BB * sizeof(float));

  normalize_split<<<BB, 256, 0, stream>>>(feat, neg, fh, fl, nh, nl);
  sim_gemm<<<dim3(BB / 16, BB / 256), 256, 0, stream>>>(fh, fl, nh, nl, S);
  row_loss<<<BB, 128, 0, stream>>>(S, neg_idx, pp);
  final_reduce<<<1, 256, 0, stream>>>(pp, out);
}
